// DGMNet_51685636440754
// MI455X (gfx1250) — matrix operand order repaired and checked by reading, compiled, not run on hardware
//
#include <hip/hip_runtime.h>

// DGMNet forward, MI455X / gfx1250, wave32, bf16 WMMA (v_wmma_f32_16x16x32_bf16).
// ~1.65 TFLOP of 1024-K GEMMs fused with gate epilogues; activations bf16 so
// they stream through the 192MB L2. Register double-buffered k-loop
// (removes s_wait_loadcnt 0x0 stalls between WMMAs), async-to-LDS x staging.

#define HDIM 1024
#define BATCHN 65536

typedef __attribute__((ext_vector_type(8)))  __bf16 bf16x8;
typedef __attribute__((ext_vector_type(16))) __bf16 bf16x16;
typedef __attribute__((ext_vector_type(8)))  float  f32x8;
typedef __attribute__((ext_vector_type(4)))  int    v4i;

#if defined(__has_builtin)
#if __has_builtin(__builtin_amdgcn_global_load_async_to_lds_b128) && \
    __has_builtin(__builtin_amdgcn_s_wait_asynccnt)
#define USE_ASYNC_LDS 1
#endif
#endif
#ifndef USE_ASYNC_LDS
#define USE_ASYNC_LDS 0
#endif

__device__ __forceinline__ unsigned short f2bf(float f) {
  unsigned u = __float_as_uint(f);
  unsigned r = u + 0x7FFFu + ((u >> 16) & 1u);   // round-to-nearest-even
  return (unsigned short)(r >> 16);
}
__device__ __forceinline__ float bf2f(unsigned short h) {
  return __uint_as_float(((unsigned)h) << 16);
}

__device__ __forceinline__ bf16x16 concat8(bf16x8 lo, bf16x8 hi) {
  return __builtin_shufflevector(lo, hi, 0,1,2,3,4,5,6,7,8,9,10,11,12,13,14,15);
}

// A fragment (16x32 bf16, MxK): lane l holds row M = l&15; lane group l>>4
// selects K-phase. Per ISA table: VGPR0-3 = K {kh*8+0..7}, VGPR4-7 = K {16+kh*8+0..7}.
__device__ __forceinline__ bf16x16 load_a_frag(const unsigned short* __restrict__ base,
                                               int m, int kc, int khalf) {
  const unsigned short* p = base + (size_t)m * HDIM + kc * 32 + khalf * 8;
  bf16x8 lo = *(const bf16x8*)p;          // K = kc*32 + kh*8 + [0..7]
  bf16x8 hi = *(const bf16x8*)(p + 16);   // K = kc*32 + 16 + kh*8 + [0..7]
  return concat8(lo, hi);
}

// B fragment (32x16 bf16, KxN), pre-packed: 32 lanes x 16 bf16 contiguous.
__device__ __forceinline__ bf16x16 load_b_frag(const unsigned short* __restrict__ pack,
                                               int nt, int kc, int lane) {
  const unsigned short* p = pack + (((size_t)nt * 32 + kc) * 32 + lane) * 16;
  bf16x8 lo = *(const bf16x8*)p;
  bf16x8 hi = *(const bf16x8*)(p + 8);
  return concat8(lo, hi);
}

__device__ __forceinline__ f32x8 wmma_bf16(bf16x16 a, bf16x16 b, f32x8 c) {
  return __builtin_amdgcn_wmma_f32_16x16x32_bf16(false, a, false, b, (short)0, c,
                                                 false, false);
}

// Stage `chunks` 16-byte blocks of x into LDS. Async DMA path on gfx1250
// (GLOBAL_LOAD_ASYNC_TO_LDS_B128, tracked by ASYNCcnt), plain stores otherwise.
__device__ __forceinline__ void stage_x(const float* __restrict__ xg, float* xs,
                                        int chunks, int tid) {
#if USE_ASYNC_LDS
  typedef __attribute__((address_space(1))) v4i as1_v4i;
  typedef __attribute__((address_space(3))) v4i as3_v4i;
  for (int c = tid; c < chunks; c += 256) {
    as1_v4i* gp = (as1_v4i*)(unsigned long long)((const char*)xg + (size_t)c * 16);
    as3_v4i* lp = (as3_v4i*)(unsigned int)(unsigned long long)((char*)xs + (size_t)c * 16);
    __builtin_amdgcn_global_load_async_to_lds_b128(gp, lp, 0, 0);
  }
  __builtin_amdgcn_s_wait_asynccnt(0);
#else
  for (int c = tid; c < chunks; c += 256) {
    float4 v = *(const float4*)((const char*)xg + (size_t)c * 16);
    *(float4*)((char*)xs + (size_t)c * 16) = v;
  }
#endif
  __syncthreads();
}

// ---- repack fp32 weight W[N=1024][K=1024] into bf16 B-fragment order -------
// B layout (dense 16-bit, 32x16): lane l -> column n = l&15; k = (l>>4)*8 + (e&7) + (e>>3)*16.
__global__ __launch_bounds__(256) void k_pack(const float* __restrict__ W,
                                              unsigned short* __restrict__ P) {
  unsigned idx = blockIdx.x * 256u + threadIdx.x;   // 0 .. 2^20-1
  int e    = idx & 15;
  int lane = (idx >> 4) & 31;
  int kc   = (idx >> 9) & 31;
  int nt   = idx >> 14;                              // 0..63
  int n = nt * 16 + (lane & 15);
  int k = kc * 32 + (lane >> 4) * 8 + (e & 7) + (e >> 3) * 16;
  P[idx] = f2bf(W[(size_t)n * HDIM + k]);
}

// ---- S1 = x @ Sw^T + Sw_b  (K=16, VALU) ------------------------------------
__global__ __launch_bounds__(256) void k_init_s1(const float* __restrict__ x,
                                                 const float* __restrict__ Sw_w,
                                                 const float* __restrict__ Sw_b,
                                                 unsigned short* __restrict__ S) {
  size_t idx = (size_t)blockIdx.x * 256 + threadIdx.x;
  int n = (int)(idx & (HDIM - 1));
  size_t b = idx >> 10;
  float s = Sw_b[n];
#pragma unroll
  for (int k = 0; k < 16; ++k)
    s = __builtin_fmaf(x[b * 16 + k], Sw_w[(size_t)n * 16 + k], s);
  S[idx] = f2bf(s);
}

// ---- gate R: P = S@Wg^T + bg; R = tanh(ux_r + P); SR = S*R -----------------
// block = 128x128 output tile, 8 waves (2x4), wave = 64x32 (4x2 WMMA tiles)
__global__ __launch_bounds__(256) void k_gate_r(
    const unsigned short* __restrict__ S, const unsigned short* __restrict__ WgPack,
    const float* __restrict__ x,
    const float* __restrict__ Ur_w, const float* __restrict__ Ur_b,
    const float* __restrict__ Wg_b,
    unsigned short* __restrict__ SR, unsigned short* __restrict__ wgS1, int saveP) {
  __shared__ float xs[128][16];
  const int tid = threadIdx.x, wave = tid >> 5, lane = tid & 31;
  const int mblock = blockIdx.x * 128, nblock = blockIdx.y * 128;
  stage_x(x + (size_t)mblock * 16, &xs[0][0], 128 * 16 * 4 / 16, tid);

  const int mwave = mblock + (wave & 1) * 64;
  const int nwave = nblock + (wave >> 1) * 32;
  const int lhalf = lane >> 4, lmod = lane & 15;
  const int nt0 = nwave >> 4;

  f32x8 acc[4][2];
  const f32x8 zero = {0.f,0.f,0.f,0.f,0.f,0.f,0.f,0.f};
#pragma unroll
  for (int i = 0; i < 4; ++i) { acc[i][0] = zero; acc[i][1] = zero; }

  // double-buffered fragments; fully unrolled k-loop so buffer index is static
  bf16x16 aB[2][4], b0B[2], b1B[2];
  b0B[0] = load_b_frag(WgPack, nt0, 0, lane);
  b1B[0] = load_b_frag(WgPack, nt0 + 1, 0, lane);
#pragma unroll
  for (int i = 0; i < 4; ++i)
    aB[0][i] = load_a_frag(S, mwave + i * 16 + lmod, 0, lhalf);

#pragma unroll
  for (int kc = 0; kc < 32; ++kc) {
    const int cur = kc & 1, nxt = cur ^ 1;
    if (kc + 1 < 32) {
      b0B[nxt] = load_b_frag(WgPack, nt0, kc + 1, lane);
      b1B[nxt] = load_b_frag(WgPack, nt0 + 1, kc + 1, lane);
#pragma unroll
      for (int i = 0; i < 4; ++i)
        aB[nxt][i] = load_a_frag(S, mwave + i * 16 + lmod, kc + 1, lhalf);
      if (kc + 2 < 32)
        __builtin_prefetch(WgPack + (((size_t)nt0 * 32 + kc + 2) * 32 + lane) * 16, 0, 3);
    }
#pragma unroll
    for (int i = 0; i < 4; ++i) {
      acc[i][0] = wmma_bf16(aB[cur][i], b0B[cur], acc[i][0]);
      acc[i][1] = wmma_bf16(aB[cur][i], b1B[cur], acc[i][1]);
    }
  }

#pragma unroll
  for (int j = 0; j < 2; ++j) {
    const int n = nwave + j * 16 + lmod;
    float u[16];
#pragma unroll
    for (int k = 0; k < 16; ++k) u[k] = Ur_w[(size_t)n * 16 + k];
    const float urb = Ur_b[n], bg = Wg_b[n];
#pragma unroll
    for (int i = 0; i < 4; ++i) {
#pragma unroll
      for (int r = 0; r < 8; ++r) {
        const int m = mwave + i * 16 + lhalf * 8 + r, ml = m - mblock;
        float P = acc[i][j][r] + bg;
        float ux = urb;
#pragma unroll
        for (int k = 0; k < 16; ++k) ux = __builtin_fmaf(xs[ml][k], u[k], ux);
        float R = tanhf(ux + P);
        size_t idx = (size_t)m * HDIM + n;
        SR[idx] = f2bf(bf2f(S[idx]) * R);
        if (saveP) wgS1[idx] = f2bf(P);
      }
    }
  }
}

// ---- gates Z,H,G + combine: dual GEMM (S@Wz^T, SR@Wg^T) --------------------
// block = 64x128, 8 waves (2x4), wave = 32x32 (2x2 tiles, two accumulator sets)
__global__ __launch_bounds__(256) void k_gate_zh(
    const unsigned short* __restrict__ S, const unsigned short* __restrict__ SR,
    const unsigned short* __restrict__ WgPack, const unsigned short* __restrict__ WzPack,
    const float* __restrict__ x,
    const float* __restrict__ Uz_w, const float* __restrict__ Uz_b,
    const float* __restrict__ Ug_w, const float* __restrict__ Ug_b,
    const float* __restrict__ Uh_w, const float* __restrict__ Uh_b,
    const float* __restrict__ Wz_b, const float* __restrict__ Wg_b,
    const unsigned short* __restrict__ wgS1,
    unsigned short* __restrict__ Snext, unsigned short* __restrict__ Raw, int writeRaw) {
  __shared__ float xs[64][16];
  const int tid = threadIdx.x, wave = tid >> 5, lane = tid & 31;
  const int mblock = blockIdx.x * 64, nblock = blockIdx.y * 128;
  stage_x(x + (size_t)mblock * 16, &xs[0][0], 64 * 16 * 4 / 16, tid);

  const int mwave = mblock + (wave & 1) * 32;
  const int nwave = nblock + (wave >> 1) * 32;
  const int lhalf = lane >> 4, lmod = lane & 15;
  const int nt0 = nwave >> 4;

  f32x8 accz[2][2], acch[2][2];
  const f32x8 zero = {0.f,0.f,0.f,0.f,0.f,0.f,0.f,0.f};
#pragma unroll
  for (int i = 0; i < 2; ++i) {
    accz[i][0] = zero; accz[i][1] = zero;
    acch[i][0] = zero; acch[i][1] = zero;
  }

  bf16x16 aS[2][2], aSR[2][2], bz0B[2], bz1B[2], bg0B[2], bg1B[2];
  bz0B[0] = load_b_frag(WzPack, nt0, 0, lane);
  bz1B[0] = load_b_frag(WzPack, nt0 + 1, 0, lane);
  bg0B[0] = load_b_frag(WgPack, nt0, 0, lane);
  bg1B[0] = load_b_frag(WgPack, nt0 + 1, 0, lane);
#pragma unroll
  for (int i = 0; i < 2; ++i) {
    aS[0][i]  = load_a_frag(S,  mwave + i * 16 + lmod, 0, lhalf);
    aSR[0][i] = load_a_frag(SR, mwave + i * 16 + lmod, 0, lhalf);
  }

#pragma unroll
  for (int kc = 0; kc < 32; ++kc) {
    const int cur = kc & 1, nxt = cur ^ 1;
    if (kc + 1 < 32) {
      bz0B[nxt] = load_b_frag(WzPack, nt0, kc + 1, lane);
      bz1B[nxt] = load_b_frag(WzPack, nt0 + 1, kc + 1, lane);
      bg0B[nxt] = load_b_frag(WgPack, nt0, kc + 1, lane);
      bg1B[nxt] = load_b_frag(WgPack, nt0 + 1, kc + 1, lane);
#pragma unroll
      for (int i = 0; i < 2; ++i) {
        aS[nxt][i]  = load_a_frag(S,  mwave + i * 16 + lmod, kc + 1, lhalf);
        aSR[nxt][i] = load_a_frag(SR, mwave + i * 16 + lmod, kc + 1, lhalf);
      }
      if (kc + 2 < 32) {
        __builtin_prefetch(WzPack + (((size_t)nt0 * 32 + kc + 2) * 32 + lane) * 16, 0, 3);
        __builtin_prefetch(WgPack + (((size_t)nt0 * 32 + kc + 2) * 32 + lane) * 16, 0, 3);
      }
    }
#pragma unroll
    for (int i = 0; i < 2; ++i) {
      accz[i][0] = wmma_bf16(aS[cur][i],  bz0B[cur], accz[i][0]);
      accz[i][1] = wmma_bf16(aS[cur][i],  bz1B[cur], accz[i][1]);
      acch[i][0] = wmma_bf16(aSR[cur][i], bg0B[cur], acch[i][0]);
      acch[i][1] = wmma_bf16(aSR[cur][i], bg1B[cur], acch[i][1]);
    }
  }

#pragma unroll
  for (int j = 0; j < 2; ++j) {
    const int n = nwave + j * 16 + lmod;
    float uz[16], ug[16], uh[16];
#pragma unroll
    for (int k = 0; k < 16; ++k) {
      uz[k] = Uz_w[(size_t)n * 16 + k];
      ug[k] = Ug_w[(size_t)n * 16 + k];
      uh[k] = Uh_w[(size_t)n * 16 + k];
    }
    const float uzb = Uz_b[n], ugb = Ug_b[n], uhb = Uh_b[n];
    const float bz = Wz_b[n], bg = Wg_b[n];
#pragma unroll
    for (int i = 0; i < 2; ++i) {
#pragma unroll
      for (int r = 0; r < 8; ++r) {
        const int m = mwave + i * 16 + lhalf * 8 + r, ml = m - mblock;
        float dz = uzb, dg = ugb, dh = uhb;
#pragma unroll
        for (int k = 0; k < 16; ++k) {
          float xv = xs[ml][k];
          dz = __builtin_fmaf(xv, uz[k], dz);
          dg = __builtin_fmaf(xv, ug[k], dg);
          dh = __builtin_fmaf(xv, uh[k], dh);
        }
        size_t idx = (size_t)m * HDIM + n;
        float Zv = tanhf(dz + accz[i][j][r] + bz);
        float Hv = tanhf(dh + acch[i][j][r] + bg);
        float Gv = tanhf(dg + bf2f(wgS1[idx]));
        float Sval = bf2f(S[idx]);
        float out = (1.f - Gv) * Hv + Zv * Sval;
        Snext[idx] = f2bf(tanhf(out));
        if (writeRaw) Raw[idx] = f2bf(out);   // aliases wgS1; per-thread RAW-safe
      }
    }
  }
}

// ---- final: y[b] = sum_n Raw[b][n]*out_w[n] + out_b ------------------------
__global__ __launch_bounds__(256) void k_out(const unsigned short* __restrict__ Raw,
                                             const float* __restrict__ out_w,
                                             const float* __restrict__ out_b,
                                             float* __restrict__ y) {
  const int tid = threadIdx.x, lane = tid & 31;
  const size_t b = (size_t)blockIdx.x * 8 + (tid >> 5);
  const unsigned short* row = Raw + b * HDIM;
  float s = 0.f;
#pragma unroll
  for (int j = 0; j < 32; ++j) {
    int c = lane + j * 32;
    s = __builtin_fmaf(bf2f(row[c]), out_w[c], s);
  }
#pragma unroll
  for (int off = 16; off; off >>= 1) s += __shfl_xor(s, off, 32);
  if (lane == 0) y[b] = s + out_b[0];
}

extern "C" void kernel_launch(void* const* d_in, const int* in_sizes, int n_in,
                              void* d_out, int out_size, void* d_ws, size_t ws_size,
                              hipStream_t stream) {
  const float* x    = (const float*)d_in[0];
  const float* Sw_w = (const float*)d_in[1];
  const float* Sw_b = (const float*)d_in[2];
  const float* Uz_w = (const float*)d_in[3];
  const float* Uz_b = (const float*)d_in[4];
  const float* Wz_w = (const float*)d_in[5];
  const float* Wz_b = (const float*)d_in[6];
  const float* Ug_w = (const float*)d_in[7];
  const float* Ug_b = (const float*)d_in[8];
  const float* Wg_w = (const float*)d_in[9];
  const float* Wg_b = (const float*)d_in[10];
  const float* Ur_w = (const float*)d_in[11];
  const float* Ur_b = (const float*)d_in[12];
  const float* Uh_w = (const float*)d_in[13];
  const float* Uh_b = (const float*)d_in[14];
  const float* out_w = (const float*)d_in[15];
  const float* out_b = (const float*)d_in[16];

  const size_t MB = 1u << 20;
  const size_t ACT = (size_t)BATCHN * HDIM * 2;   // 128 MB bf16
  char* ws = (char*)d_ws;
  unsigned short* WgPack = (unsigned short*)(ws);
  unsigned short* WzPack = (unsigned short*)(ws + 2 * MB);
  unsigned short* Sa     = (unsigned short*)(ws + 4 * MB);
  unsigned short* Sb     = (unsigned short*)(ws + 4 * MB + ACT);
  unsigned short* SR     = (unsigned short*)(ws + 4 * MB + 2 * ACT);
  unsigned short* WG1    = (unsigned short*)(ws + 4 * MB + 3 * ACT);
  unsigned short* Raw    = WG1;  // only written at last layer (read-before-write per elem)

  k_pack<<<4096, 256, 0, stream>>>(Wg_w, WgPack);
  k_pack<<<4096, 256, 0, stream>>>(Wz_w, WzPack);
  k_init_s1<<<(BATCHN * HDIM) / 256, 256, 0, stream>>>(x, Sw_w, Sw_b, Sa);

  unsigned short* Scur = Sa;
  unsigned short* Snxt = Sb;
  for (int layer = 0; layer < 4; ++layer) {
    k_gate_r<<<dim3(BATCHN / 128, HDIM / 128), 256, 0, stream>>>(
        Scur, WgPack, x, Ur_w, Ur_b, Wg_b, SR, WG1, layer == 0 ? 1 : 0);
    k_gate_zh<<<dim3(BATCHN / 64, HDIM / 128), 256, 0, stream>>>(
        Scur, SR, WgPack, WzPack, x, Uz_w, Uz_b, Ug_w, Ug_b, Uh_w, Uh_b,
        Wz_b, Wg_b, WG1, Snxt, Raw, layer == 3 ? 1 : 0);
    unsigned short* t = Scur; Scur = Snxt; Snxt = t;
  }
  k_out<<<BATCHN / 8, 256, 0, stream>>>(Raw, out_w, out_b, (float*)d_out);
}
